// GQAParallelAttention_74217034875321
// MI455X (gfx1250) — matrix operand order repaired and checked by reading, not compiled, not run on hardware
//
#include <hip/hip_runtime.h>
#include <hip/hip_bf16.h>

typedef __attribute__((ext_vector_type(16))) __bf16 v16bf;
typedef __attribute__((ext_vector_type(8)))  __bf16 v8bf;
typedef __attribute__((ext_vector_type(4)))  __bf16 v4bf;
typedef __attribute__((ext_vector_type(8)))  float  v8f;
typedef int v4i_ __attribute__((vector_size(16)));

constexpr int B_   = 2;
constexpr int S_   = 2048;
constexpr int H_   = 2048;
constexpr int NH_  = 16;
constexpr int NKV_ = 4;
constexpr int HD_  = 128;
constexpr int KV_  = NKV_ * HD_;   // 512
constexpr float SCALE_ = 0.08838834764831845f;   // 1/sqrt(128)

#if defined(__gfx1250__) && __has_builtin(__builtin_amdgcn_global_load_async_to_lds_b128)
#define HAVE_ASYNC_LDS 1
#else
#define HAVE_ASYNC_LDS 0
#endif

// ---------------------------------------------------------------------------
// f32 -> bf16 bulk conversion (4 elements / thread, 16B reads, 8B writes)
// ---------------------------------------------------------------------------
__global__ __launch_bounds__(256)
void cvt_f32_bf16(const float* __restrict__ src, __bf16* __restrict__ dst, size_t n) {
    size_t i = ((size_t)blockIdx.x * 256 + threadIdx.x) * 4;
    if (i >= n) return;
    float4 v = *(const float4*)&src[i];
    v4bf o;
    o[0] = (__bf16)v.x; o[1] = (__bf16)v.y; o[2] = (__bf16)v.z; o[3] = (__bf16)v.w;
    *(v4bf*)&dst[i] = o;
}

// ---------------------------------------------------------------------------
// V cache f32 [B,S,NKV,HD] -> bf16 transposed [B,NKV,HD,S] (keys contiguous)
// ---------------------------------------------------------------------------
__global__ __launch_bounds__(256)
void transpose_v_bf16(const float* __restrict__ v, __bf16* __restrict__ vt) {
    size_t idx = (size_t)blockIdx.x * 256 + threadIdx.x;
    size_t total = (size_t)B_ * NKV_ * HD_ * S_;
    if (idx >= total) return;
    int s = (int)(idx % S_);
    size_t t = idx / S_;
    int d  = (int)(t % HD_);
    size_t t2 = t / HD_;
    int kv = (int)(t2 % NKV_);
    int b  = (int)(t2 / NKV_);
    vt[idx] = (__bf16)v[((size_t)(b * S_ + s) * NKV_ + kv) * HD_ + d];
}

// ---------------------------------------------------------------------------
// GEMM: C[M,N] = A[M,K] * B[K,N]; A,B bf16, C f32. 128 thr (4 waves),
// 128x64 block tile (32x64 per wave), K-step 32. A tile staged to LDS with
// async copies (ASYNCcnt path) when available; B tile transposed to [n][k]
// so all WMMA fragments are contiguous 16B ds_load_b128 chunks.
// ---------------------------------------------------------------------------
__global__ __launch_bounds__(128)
void gemm_bf16_wmma(const __bf16* __restrict__ A, const __bf16* __restrict__ Bm,
                    float* __restrict__ C, int M, int N, int K) {
    __shared__ alignas(32) __bf16 As[128][32];   // [m][k]
    __shared__ alignas(32) __bf16 Bst[64][32];   // [n][k]

    const int tid   = threadIdx.x;
    const int wave  = tid >> 5;
    const int lane  = tid & 31;
    const int lmod  = lane & 15;
    const int lhalf = lane >> 4;

    const int ntile = N >> 6;
    const int m0 = (blockIdx.x / ntile) * 128;
    const int n0 = (blockIdx.x % ntile) * 64;

    v8f zero = {};
    v8f acc[2][4];
    #pragma unroll
    for (int mt = 0; mt < 2; ++mt)
        #pragma unroll
        for (int nt = 0; nt < 4; ++nt) acc[mt][nt] = zero;

    for (int k0 = 0; k0 < K; k0 += 32) {
        // A tile: 128x32 bf16 = 512 16B chunks, 4 per thread
        #pragma unroll
        for (int i = 0; i < 4; ++i) {
            int idx = tid * 4 + i;                 // 0..511
            int r = idx >> 2, c = (idx & 3) * 8;   // 4 chunks per 32-elem row
            const __bf16* gsrc = &A[(size_t)(m0 + r) * K + k0 + c];
#if HAVE_ASYNC_LDS
            __builtin_amdgcn_global_load_async_to_lds_b128(
                (__attribute__((address_space(1))) v4i_*)gsrc,
                (__attribute__((address_space(3))) v4i_*)&As[r][c],
                0, 0);
#else
            *(v8bf*)&As[r][c] = *(const v8bf*)gsrc;
#endif
        }
        // B tile: 32x64 bf16 read coalesced, scattered transposed into Bst[n][k]
        #pragma unroll
        for (int i = 0; i < 2; ++i) {
            int idx = tid * 2 + i;
            int r = idx >> 3, c = (idx & 7) * 8;   // 8 chunks per 64-elem row
            v8bf t8 = *(const v8bf*)&Bm[(size_t)(k0 + r) * N + n0 + c];
            #pragma unroll
            for (int j = 0; j < 8; ++j) Bst[c + j][r] = t8[j];
        }
        // prefetch next K tile (global_prefetch_b8)
        if (k0 + 32 < K) {
            __builtin_prefetch(&A[(size_t)(m0 + tid) * K + k0 + 32], 0, 1);
            __builtin_prefetch(&Bm[(size_t)(k0 + 32 + (tid & 31)) * N + n0], 0, 1);
        }
#if HAVE_ASYNC_LDS
#if __has_builtin(__builtin_amdgcn_s_wait_asynccnt)
        __builtin_amdgcn_s_wait_asynccnt(0);
#else
        asm volatile("s_wait_asynccnt 0" ::: "memory");
#endif
#endif
        __syncthreads();

        // B fragments (shared across both A fragments of this wave)
        v16bf bfr[4];
        #pragma unroll
        for (int nt = 0; nt < 4; ++nt) {
            const int bc = nt * 16 + lmod;
            v8bf blo = *(const v8bf*)&Bst[bc][lhalf * 8];
            v8bf bhi = *(const v8bf*)&Bst[bc][16 + lhalf * 8];
            #pragma unroll
            for (int i = 0; i < 8; ++i) { bfr[nt][i] = blo[i]; bfr[nt][i + 8] = bhi[i]; }
        }
        #pragma unroll
        for (int mt = 0; mt < 2; ++mt) {
            const int am = wave * 32 + mt * 16 + lmod;
            v8bf alo = *(const v8bf*)&As[am][lhalf * 8];
            v8bf ahi = *(const v8bf*)&As[am][16 + lhalf * 8];
            v16bf af;
            #pragma unroll
            for (int i = 0; i < 8; ++i) { af[i] = alo[i]; af[i + 8] = ahi[i]; }
            #pragma unroll
            for (int nt = 0; nt < 4; ++nt)
                acc[mt][nt] = __builtin_amdgcn_wmma_f32_16x16x32_bf16(
                    false, af, false, bfr[nt], (short)0, acc[mt][nt], false, false);
        }
        __syncthreads();
    }

    #pragma unroll
    for (int mt = 0; mt < 2; ++mt)
        #pragma unroll
        for (int nt = 0; nt < 4; ++nt) {
            const int col = n0 + nt * 16 + lmod;
            #pragma unroll
            for (int j = 0; j < 8; ++j) {
                const int row = m0 + wave * 32 + mt * 16 + j + lhalf * 8;
                C[(size_t)row * N + col] = acc[mt][nt][j];
            }
        }
}

// ---------------------------------------------------------------------------
// RoPE in-place on q [B,S,NH,HD] and cache_k [B,S,NKV,HD] (f32).
// ---------------------------------------------------------------------------
__global__ __launch_bounds__(256)
void rope_kernel(float* __restrict__ q, float* __restrict__ kc,
                 const int* __restrict__ pos_ids) {
    const int half = HD_ / 2;
    size_t idx = (size_t)blockIdx.x * blockDim.x + threadIdx.x;
    size_t total = (size_t)B_ * S_ * (NH_ + NKV_) * half;
    if (idx >= total) return;
    int d = (int)(idx % half);
    size_t t = idx / half;
    int h = (int)(t % (NH_ + NKV_));
    size_t bs = t / (NH_ + NKV_);

    float p = (float)pos_ids[bs];
    float freq = exp2f(-((float)d / (float)half) * 19.931568569324174f); // log2(1e6)
    float sn, cs;
    __sincosf(p * freq, &sn, &cs);

    float* base = (h < NH_) ? (q  + ((bs * NH_  + h)         * (size_t)HD_))
                            : (kc + ((bs * NKV_ + (h - NH_)) * (size_t)HD_));
    float x1 = base[d], x2 = base[d + half];
    base[d]        = x1 * cs - x2 * sn;
    base[d + half] = x1 * sn + x2 * cs;
}

// ---------------------------------------------------------------------------
// Flash attention: one wave per (b, h, 16-row q tile). bf16 operands,
// all fragment loads are 16B vector loads. Output attn bf16.
//   q  : [B,S,NH,HD]    bf16 (roped)
//   kc : [B,S,NKV,HD]   bf16 (roped)
//   vt : [B,NKV,HD,S]   bf16 (transposed)
// ---------------------------------------------------------------------------
__global__ __launch_bounds__(32)
void flash_attn_wmma(const __bf16* __restrict__ q, const __bf16* __restrict__ kc,
                     const __bf16* __restrict__ vt, __bf16* __restrict__ attn_out) {
    __shared__ __bf16 Plds[16][32];

    const int lane  = threadIdx.x & 31;
    const int lmod  = lane & 15;
    const int lhalf = lane >> 4;

    const int qtiles = S_ / 16;
    const int qt = blockIdx.x % qtiles;
    const int h  = (blockIdx.x / qtiles) % NH_;
    const int b  = blockIdx.x / (qtiles * NH_);
    const int kvh = h >> 2;            // NH/NKV = 4
    const int q0 = qt * 16;

    // Q fragments (A-layout): two 16B loads per 32-dim chunk
    v16bf qf[4];
    {
        const size_t qbase = ((size_t)(b * S_ + q0 + lmod) * NH_ + h) * HD_;
        #pragma unroll
        for (int c = 0; c < 4; ++c) {
            v8bf lo = *(const v8bf*)&q[qbase + c * 32 + lhalf * 8];
            v8bf hi = *(const v8bf*)&q[qbase + c * 32 + 16 + lhalf * 8];
            #pragma unroll
            for (int i = 0; i < 8; ++i) { qf[c][i] = lo[i]; qf[c][i + 8] = hi[i]; }
        }
    }

    v8f zero = {};
    v8f O[8];
    #pragma unroll
    for (int t = 0; t < 8; ++t) O[t] = zero;
    float m_i[8], l_i[8];
    #pragma unroll
    for (int j = 0; j < 8; ++j) { m_i[j] = -3.0e38f; l_i[j] = 0.0f; }

    const size_t kvrow_base = (size_t)b * S_ * KV_ + (size_t)kvh * HD_;    // + s*KV_
    const size_t vtbase     = ((size_t)(b * NKV_ + kvh) * HD_) * S_;       // + d*S_ + s

    const int kb_end = (q0 + 15) >> 5;
    for (int kb = 0; kb <= kb_end; ++kb) {
        // ---- scores: two 16x16 accumulators covering 32 keys ----
        v8f s0 = zero, s1 = zero;
        const int key0 = kb * 32 + lmod;
        #pragma unroll
        for (int c = 0; c < 4; ++c) {
            const size_t b0 = kvrow_base + (size_t)key0 * KV_        + c * 32 + lhalf * 8;
            const size_t b1 = kvrow_base + (size_t)(key0 + 16) * KV_ + c * 32 + lhalf * 8;
            v8bf k0lo = *(const v8bf*)&kc[b0];
            v8bf k0hi = *(const v8bf*)&kc[b0 + 16];
            v8bf k1lo = *(const v8bf*)&kc[b1];
            v8bf k1hi = *(const v8bf*)&kc[b1 + 16];
            v16bf kb0, kb1;
            #pragma unroll
            for (int i = 0; i < 8; ++i) {
                kb0[i] = k0lo[i]; kb0[i + 8] = k0hi[i];
                kb1[i] = k1lo[i]; kb1[i + 8] = k1hi[i];
            }
            s0 = __builtin_amdgcn_wmma_f32_16x16x32_bf16(false, qf[c], false, kb0,
                                                         (short)0, s0, false, false);
            s1 = __builtin_amdgcn_wmma_f32_16x16x32_bf16(false, qf[c], false, kb1,
                                                         (short)0, s1, false, false);
        }

        // ---- scale + causal mask + online softmax ----
        #pragma unroll
        for (int j = 0; j < 8; ++j) {
            const int row_g = q0 + j + lhalf * 8;
            const int col0 = kb * 32 + lmod;
            const int col1 = col0 + 16;
            float a0 = s0[j] * SCALE_; if (col0 > row_g) a0 = -3.0e38f;
            float a1 = s1[j] * SCALE_; if (col1 > row_g) a1 = -3.0e38f;

            float mx = fmaxf(a0, a1);
            #pragma unroll
            for (int msk = 1; msk < 16; msk <<= 1) mx = fmaxf(mx, __shfl_xor(mx, msk, 32));
            float mnew  = fmaxf(m_i[j], mx);
            float alpha = __expf(m_i[j] - mnew);
            float p0 = __expf(a0 - mnew);
            float p1 = __expf(a1 - mnew);
            float rs = p0 + p1;
            #pragma unroll
            for (int msk = 1; msk < 16; msk <<= 1) rs += __shfl_xor(rs, msk, 32);
            l_i[j] = l_i[j] * alpha + rs;
            m_i[j] = mnew;
            #pragma unroll
            for (int t = 0; t < 8; ++t) O[t][j] *= alpha;
            Plds[j + lhalf * 8][lmod]      = (__bf16)p0;
            Plds[j + lhalf * 8][lmod + 16] = (__bf16)p1;
        }
        asm volatile("s_wait_dscnt 0" ::: "memory");

        // ---- P as A-fragment ----
        v16bf pa;
        #pragma unroll
        for (int i = 0; i < 16; ++i) {
            int kk = (i < 8 ? i : i + 8) + lhalf * 8;
            pa[i] = Plds[lmod][kk];
        }
        asm volatile("" ::: "memory");

        // ---- O += P x V (V pre-transposed: keys contiguous per dim row) ----
        #pragma unroll
        for (int t = 0; t < 8; ++t) {
            const size_t vb0 = vtbase + (size_t)(t * 16 + lmod) * S_ + kb * 32 + lhalf * 8;
            v8bf vlo = *(const v8bf*)&vt[vb0];
            v8bf vhi = *(const v8bf*)&vt[vb0 + 16];
            v16bf vb;
            #pragma unroll
            for (int i = 0; i < 8; ++i) { vb[i] = vlo[i]; vb[i + 8] = vhi[i]; }
            O[t] = __builtin_amdgcn_wmma_f32_16x16x32_bf16(false, pa, false, vb,
                                                           (short)0, O[t], false, false);
        }
    }

    // ---- normalize and store attn (bf16) [B,S,NH,HD] ----
    #pragma unroll
    for (int j = 0; j < 8; ++j) {
        const int row_g = q0 + j + lhalf * 8;
        const float inv_l = 1.0f / l_i[j];
        #pragma unroll
        for (int t = 0; t < 8; ++t)
            attn_out[((size_t)(b * S_ + row_g) * NH_ + h) * HD_ + t * 16 + lmod] =
                (__bf16)(O[t][j] * inv_l);
    }
}

// ---------------------------------------------------------------------------
extern "C" void kernel_launch(void* const* d_in, const int* in_sizes, int n_in,
                              void* d_out, int out_size, void* d_ws, size_t ws_size,
                              hipStream_t stream) {
    const float* hs  = (const float*)d_in[0];
    const int*   pos = (const int*)  d_in[2];
    const float* Wq  = (const float*)d_in[3];
    const float* Wk  = (const float*)d_in[4];
    const float* Wv  = (const float*)d_in[5];
    const float* Wo  = (const float*)d_in[6];

    float* out     = (float*)d_out;
    float* cache_k = out + (size_t)B_ * S_ * H_;
    float* cache_v = cache_k + (size_t)B_ * S_ * KV_;

    const size_t n_hs = (size_t)B_ * S_ * H_;      // 8388608
    const size_t n_wq = (size_t)H_ * H_;           // 4194304
    const size_t n_wk = (size_t)H_ * KV_;          // 1048576
    const size_t n_kv = (size_t)B_ * S_ * KV_;     // 2097152

    // workspace layout (bf16 buffers first, then f32 q)
    __bf16* hs_bf   = (__bf16*)d_ws;
    __bf16* wq_bf   = hs_bf + n_hs;
    __bf16* wk_bf   = wq_bf + n_wq;
    __bf16* wv_bf   = wk_bf + n_wk;
    __bf16* wo_bf   = wv_bf + n_wk;
    float*  q_f32   = (float*)(wo_bf + n_wq);
    __bf16* q_bf    = (__bf16*)(q_f32 + n_hs);
    __bf16* k_bf    = q_bf + n_hs;
    __bf16* vt_bf   = k_bf + n_kv;
    __bf16* attn_bf = vt_bf + n_kv;

    const int M = B_ * S_;   // 4096

    auto cvt = [&](const float* s, __bf16* d, size_t n) {
        cvt_f32_bf16<<<(unsigned)((n / 4 + 255) / 256), 256, 0, stream>>>(s, d, n);
    };

    // 1) precision conversion of all GEMM operands
    cvt(hs, hs_bf, n_hs);
    cvt(Wq, wq_bf, n_wq);
    cvt(Wk, wk_bf, n_wk);
    cvt(Wv, wv_bf, n_wk);
    cvt(Wo, wo_bf, n_wq);

    // 2) projections (K/V land in the cache output regions, f32)
    gemm_bf16_wmma<<<(M / 128) * (H_  / 64), 128, 0, stream>>>(hs_bf, wq_bf, q_f32,   M, H_,  H_);
    gemm_bf16_wmma<<<(M / 128) * (KV_ / 64), 128, 0, stream>>>(hs_bf, wk_bf, cache_k, M, KV_, H_);
    gemm_bf16_wmma<<<(M / 128) * (KV_ / 64), 128, 0, stream>>>(hs_bf, wv_bf, cache_v, M, KV_, H_);

    // 3) RoPE (in place, f32) on q and cache_k
    {
        size_t tot = (size_t)B_ * S_ * (NH_ + NKV_) * (HD_ / 2);
        rope_kernel<<<(unsigned)((tot + 255) / 256), 256, 0, stream>>>(q_f32, cache_k, pos);
    }

    // 4) bf16 copies for the attention kernel (q, k; V transposed)
    cvt(q_f32, q_bf, n_hs);
    cvt(cache_k, k_bf, n_kv);
    transpose_v_bf16<<<(unsigned)((n_kv + 255) / 256), 256, 0, stream>>>(cache_v, vt_bf);

    // 5) flash attention -> attn_bf [B,S,NH,HD]
    flash_attn_wmma<<<B_ * NH_ * (S_ / 16), 32, 0, stream>>>(q_bf, k_bf, vt_bf, attn_bf);

    // 6) output projection
    gemm_bf16_wmma<<<(M / 128) * (H_ / 64), 128, 0, stream>>>(attn_bf, wo_bf, out, M, H_, H_);
}
